// TransformerBlock_21715354649033
// MI455X (gfx1250) — matrix operand order repaired and checked by reading, not compiled, not run on hardware
//
#include <hip/hip_runtime.h>
#include <hip/hip_bf16.h>

// ---------------------------------------------------------------------------
// MI455X (gfx1250) transformer block, WMMA bf16 with TDM + async-LDS pipeline:
//   QKV proj -> causal attention (scores / softmax / PV) -> out proj + residual
//   -> FFN (relu) + residual.
// GEMM staging per k-stage (double buffered LDS):
//   A tile 128x32 : TENSOR_LOAD_TO_LDS (one D# descriptor, wave 0, TENSORcnt)
//   B^T tile 128x32: GLOBAL_LOAD_ASYNC_TO_LDS_B128 (all waves, ASYNCcnt)
// so the next tile's HBM/L2 traffic overlaps the current tile's 8 v_wmma ops.
// TDM issue is guarded by a readfirstlane-based *scalar* branch: TDM ignores
// EXEC, so an exec-mask guard would re-issue the DMA from every wave.
// ---------------------------------------------------------------------------

typedef __attribute__((ext_vector_type(16))) __bf16 v16bf;
typedef __attribute__((ext_vector_type(8)))  float  v8f;
typedef __attribute__((ext_vector_type(4)))  unsigned u32x4;
typedef __attribute__((ext_vector_type(8)))  int      i32x8;
typedef __attribute__((ext_vector_type(4)))  int      i32x4;

#define BM 128
#define BN 128
#define BK 32

struct GemmP {
  const __bf16* A;  long lda; long sA1, sA2;   // A [M][K], off=(z/H)*sA1+(z%H)*sA2
  const __bf16* Bt; long ldb; long sB1, sB2;   // B^T [N][K] row-major
  float* Cf; __bf16* Cb; long ldc; long sC1, sC2;
  const float* bias;                            // per output column, nullable
  const float* R; long ldr; long sR1, sR2;      // fp32 residual, nullable
  int M, N, K, Hdiv;
  float alpha;
  int relu, causal, ctrans;                     // ctrans: store Cb transposed
};

// ---- TDM: load a 2-D tile (BK cols x BM rows, bf16, row stride ld) to LDS --
__device__ __forceinline__ void tdm_load_tile(const __bf16* g, long ld,
                                              unsigned lds_off) {
  unsigned long long ga = (unsigned long long)(size_t)g;
  unsigned long long st = (unsigned long long)ld;   // dim0 stride, elements
  u32x4 g0; i32x8 g1; i32x4 gz;
  gz[0] = 0; gz[1] = 0; gz[2] = 0; gz[3] = 0;
  g0[0] = 1u;                                        // count=1, user mode
  g0[1] = lds_off;                                   // lds_addr (bytes)
  g0[2] = (unsigned)ga;                              // global_addr[31:0]
  g0[3] = (unsigned)((ga >> 32) & 0x01FFFFFFu) | (2u << 30);  // addr[56:32]|type=2
  g1[0] = (int)(1u << 16);                           // data_size=1 (2 bytes)
  g1[1] = (int)(((unsigned)st & 0xFFFFu) << 16);     // tensor_dim0[15:0]
  g1[2] = (int)((((unsigned)st >> 16) & 0xFFFFu)     // tensor_dim0[31:16]
                | ((unsigned)BM << 16));             // tensor_dim1[15:0]=128
  g1[3] = (int)((unsigned)BK << 16);                 // tile_dim0=32
  g1[4] = (int)(unsigned)BM;                         // tile_dim1=128, tile_dim2=0
  g1[5] = (int)(unsigned)st;                         // dim0_stride[31:0]
  g1[6] = (int)((unsigned)(st >> 32) & 0xFFFFu);     // dim0_stride[47:32]
  g1[7] = 0;
#if __clang_major__ >= 23
  i32x8 gz8;
  gz8[0]=0; gz8[1]=0; gz8[2]=0; gz8[3]=0; gz8[4]=0; gz8[5]=0; gz8[6]=0; gz8[7]=0;
  __builtin_amdgcn_tensor_load_to_lds(g0, g1, gz, gz, gz8, 0);
#else
  __builtin_amdgcn_tensor_load_to_lds(g0, g1, gz, gz, 0);
#endif
}

// ---- async copy: one k-stage of the B^T tile (128x32), 2 instr per thread --
__device__ __forceinline__ void issue_b_stage(const __bf16* Bt, long ldb, int n0,
                                              int k0, __bf16* bs, int tid) {
  #pragma unroll
  for (int it = 0; it < 2; ++it) {
    int c   = tid + it * 256;       // 512 chunks of 8 bf16
    int r   = c >> 2;
    int col = (c & 3) * 8;
    const void* gb = (const void*)(Bt + (long)(n0 + r) * ldb + k0 + col);
    unsigned    lb = (unsigned)(size_t)(bs + r * BK + col);
    asm volatile("global_load_async_to_lds_b128 %0, %1, off"
                 :: "v"(lb), "v"(gb) : "memory");
  }
}

__global__ __launch_bounds__(256) void gemm_bf16_wmma(GemmP p) {
  __shared__ __align__(16) __bf16 As[2][BM * BK];   // [m][k], double buffered
  __shared__ __align__(16) __bf16 Bs[2][BN * BK];   // [n][k], double buffered

  const int z  = blockIdx.z;
  const int zb = z / p.Hdiv, zh = z % p.Hdiv;
  const __bf16* A  = p.A  + (long)zb * p.sA1 + (long)zh * p.sA2;
  const __bf16* Bt = p.Bt + (long)zb * p.sB1 + (long)zh * p.sB2;
  const long coff  = (long)zb * p.sC1 + (long)zh * p.sC2;

  const int m0 = blockIdx.y * BM;
  const int n0 = blockIdx.x * BN;
  if (p.causal && n0 > m0 + BM - 1) return;   // block entirely above diagonal

  const int tid  = threadIdx.x;
  const int lane = tid & 31;
  const int wave = tid >> 5;
  const int wm = (wave & 3) * 32;   // 4 wave rows of 32
  const int wn = (wave >> 2) * 64;  // 2 wave cols of 64
  // wave id as an SGPR value -> uniform scalar branch around TDM issue
  // (TDM issues regardless of EXEC; an exec-masked guard is not enough).
  const bool w0 = (__builtin_amdgcn_readfirstlane(wave) == 0);

  v8f acc[2][4];
  #pragma unroll
  for (int i = 0; i < 2; ++i)
    #pragma unroll
    for (int j = 0; j < 4; ++j)
      #pragma unroll
      for (int e = 0; e < 8; ++e) acc[i][j][e] = 0.f;

  const __bf16* Atile = A + (long)m0 * p.lda;       // row-tile base
  const int nstages = p.K / BK;

  if (w0)
    tdm_load_tile(Atile, p.lda, (unsigned)(size_t)&As[0][0]);
  issue_b_stage(Bt, p.ldb, n0, 0, Bs[0], tid);

  for (int s = 0; s < nstages; ++s) {
    const int cur = s & 1;
    if (s + 1 < nstages) {
      if (w0)
        tdm_load_tile(Atile + (s + 1) * BK, p.lda,
                      (unsigned)(size_t)&As[cur ^ 1][0]);
      issue_b_stage(Bt, p.ldb, n0, (s + 1) * BK, Bs[cur ^ 1], tid);
      asm volatile("s_wait_asynccnt 2" ::: "memory");  // cur B stage complete
      if (w0) __builtin_amdgcn_s_wait_tensorcnt(1);    // cur A complete
    } else {
      asm volatile("s_wait_asynccnt 0" ::: "memory");
      if (w0) __builtin_amdgcn_s_wait_tensorcnt(0);
    }
    __syncthreads();

    // fragments per documented 16-bit A/B WMMA VGPR layouts
    union { v16bf v; unsigned u[8]; } fa[2], fb[4];
    #pragma unroll
    for (int i = 0; i < 2; ++i) {
      int row = wm + i * 16 + (lane & 15);
      #pragma unroll
      for (int q = 0; q < 8; ++q) {
        int kk = ((q & 3) * 2) + ((lane >> 4) * 8) + ((q >> 2) * 16);
        fa[i].u[q] = *(const unsigned*)&As[cur][row * BK + kk];
      }
    }
    #pragma unroll
    for (int j = 0; j < 4; ++j) {
      int nn = wn + j * 16 + (lane & 15);
      #pragma unroll
      for (int q = 0; q < 8; ++q) {
        int kk = ((q & 3) * 2) + ((lane >> 4) * 8) + ((q >> 2) * 16);
        fb[j].u[q] = *(const unsigned*)&Bs[cur][nn * BK + kk];
      }
    }
    #pragma unroll
    for (int i = 0; i < 2; ++i)
      #pragma unroll
      for (int j = 0; j < 4; ++j)
        acc[i][j] = __builtin_amdgcn_wmma_f32_16x16x32_bf16(
            false, fa[i].v, false, fb[j].v, (short)0, acc[i][j], false, false);
    __syncthreads();   // protect buffer cur^1 before next-iteration overwrite
  }

  // ---- epilogue: scale / bias / residual / relu; fp32 and/or bf16 stores ---
  float*  Cf = p.Cf ? p.Cf + coff : nullptr;
  __bf16* Cb = p.Cb ? p.Cb + coff : nullptr;
  const float* R = p.R ? p.R + (long)zb * p.sR1 + (long)zh * p.sR2 : nullptr;
  #pragma unroll
  for (int i = 0; i < 2; ++i) {
    #pragma unroll
    for (int j = 0; j < 4; ++j) {
      int ng = n0 + wn + j * 16 + (lane & 15);
      #pragma unroll
      for (int r = 0; r < 8; ++r) {
        int mg = m0 + wm + i * 16 + r + ((lane >> 4) * 8);
        float v = acc[i][j][r] * p.alpha;
        if (p.bias) v += p.bias[ng];
        if (R)      v += R[(long)mg * p.ldr + ng];
        if (p.relu) v = v > 0.f ? v : 0.f;
        if (Cf) Cf[(long)mg * p.ldc + ng] = v;
        if (Cb) {
          if (p.ctrans) Cb[(long)ng * p.ldc + mg] = (__bf16)v;
          else          Cb[(long)mg * p.ldc + ng] = (__bf16)v;
        }
      }
    }
  }
}

// ---------------------------------------------------------------------------
// Row-wise causal softmax: S fp32 [rows=B*H*T, T] -> P bf16 (zeros above diag)
// ---------------------------------------------------------------------------
__global__ __launch_bounds__(256) void softmax_causal(const float* S, __bf16* P, int T) {
  __shared__ float red[256];
  const long row = blockIdx.x;
  const int  q   = (int)(row % T);
  const float* s = S + row * (long)T;
  __bf16*     pr = P + row * (long)T;
  const int tid = threadIdx.x;

  float mx = -__builtin_inff();
  for (int c = tid; c <= q; c += 256) mx = fmaxf(mx, s[c]);
  red[tid] = mx; __syncthreads();
  for (int st = 128; st > 0; st >>= 1) {
    if (tid < st) red[tid] = fmaxf(red[tid], red[tid + st]);
    __syncthreads();
  }
  mx = red[0]; __syncthreads();

  float sum = 0.f;
  for (int c = tid; c <= q; c += 256) sum += __expf(s[c] - mx);
  red[tid] = sum; __syncthreads();
  for (int st = 128; st > 0; st >>= 1) {
    if (tid < st) red[tid] += red[tid + st];
    __syncthreads();
  }
  const float inv = 1.f / red[0];

  for (int c = tid; c < T; c += 256) {
    float v = (c <= q) ? __expf(s[c] - mx) * inv : 0.f;
    pr[c] = (__bf16)v;
  }
}

__global__ __launch_bounds__(256) void f32_to_bf16(const float* s, __bf16* d, long n) {
  long i = (long)blockIdx.x * 256 + threadIdx.x;
  long st = (long)gridDim.x * 256;
  for (; i < n; i += st) d[i] = (__bf16)s[i];
}

// transpose-convert: src fp32 [R][C] -> dst bf16 [C][R]
__global__ __launch_bounds__(256) void f32_to_bf16_t(const float* s, __bf16* d,
                                                     int R, int C) {
  long n = (long)R * C;
  long st = (long)gridDim.x * 256;
  for (long i = (long)blockIdx.x * 256 + threadIdx.x; i < n; i += st) {
    long r = i / C, c = i % C;
    d[c * (long)R + r] = (__bf16)s[i];
  }
}

// ---------------------------------------------------------------------------
extern "C" void kernel_launch(void* const* d_in, const int* in_sizes, int n_in,
                              void* d_out, int out_size, void* d_ws, size_t ws_size,
                              hipStream_t stream) {
  (void)in_sizes; (void)n_in; (void)out_size; (void)ws_size;
  constexpr long Kd = 1024, Hh = 4, Bb = 2, T = 2048;
  constexpr long M  = Bb * T;          // 4096 tokens
  constexpr long NQ = Kd * Hh;         // 4096
  const float scale = 0.17677669529663687f;  // 1024^-0.25

  const float* x  = (const float*)d_in[0];
  const float* Wq = (const float*)d_in[1];
  const float* Wk = (const float*)d_in[2];
  const float* Wv = (const float*)d_in[3];
  const float* Wu = (const float*)d_in[4];
  const float* bu = (const float*)d_in[5];
  const float* W1 = (const float*)d_in[6];
  const float* b1 = (const float*)d_in[7];
  const float* W2 = (const float*)d_in[8];
  const float* b2 = (const float*)d_in[9];

  // ---- workspace layout (bytes); all weight copies stored transposed ----
  char* ws = (char*)d_ws;
  __bf16* xb    = (__bf16*)(ws + 0);                 //  8 MB  [4096][1024]
  __bf16* WqbT  = (__bf16*)(ws + 8388608L);          //  8 MB  [4096][1024]
  __bf16* WkbT  = (__bf16*)(ws + 16777216L);         //  8 MB
  __bf16* WvbT  = (__bf16*)(ws + 25165824L);         //  8 MB
  __bf16* WubT  = (__bf16*)(ws + 33554432L);         //  8 MB  [1024][4096]
  __bf16* W1bT  = (__bf16*)(ws + 41943040L);         //  2 MB  [1024][1024]
  __bf16* W2bT  = (__bf16*)(ws + 44040192L);         //  2 MB
  __bf16* Q     = (__bf16*)(ws + 46137344L);         // 32 MB  [4096][4096]
  __bf16* Kp    = (__bf16*)(ws + 79691776L);         // 32 MB
  __bf16* Vt    = (__bf16*)(ws + 113246208L);        // 32 MB  [4096 d][4096 tok]
  float*  S     = (float*) (ws + 146800640L);        // 128 MB
  __bf16* P     = Q;                                 // 64 MB overlay (Q+Kp dead)
  __bf16* AO    = (__bf16*)(ws + 281018368L);        // 32 MB  attn out (pre-Wu)
  float*  xres  = (float*) (ws + 314572800L);        // 16 MB
  __bf16* xrb   = (__bf16*)(ws + 331350016L);        //  8 MB
  __bf16* h1    = (__bf16*)(ws + 339738624L);        //  8 MB

  // ---- fp32 -> bf16 conversions (weights transposed to [N][K]) ----
  f32_to_bf16  <<<1024, 256, 0, stream>>>(x,  xb,  M * Kd);
  f32_to_bf16_t<<<1024, 256, 0, stream>>>(Wq, WqbT, (int)Kd, (int)NQ);
  f32_to_bf16_t<<<1024, 256, 0, stream>>>(Wk, WkbT, (int)Kd, (int)NQ);
  f32_to_bf16_t<<<1024, 256, 0, stream>>>(Wv, WvbT, (int)Kd, (int)NQ);
  f32_to_bf16_t<<<1024, 256, 0, stream>>>(Wu, WubT, (int)NQ, (int)Kd);
  f32_to_bf16_t<<<256,  256, 0, stream>>>(W1, W1bT, (int)Kd, (int)Kd);
  f32_to_bf16_t<<<256,  256, 0, stream>>>(W2, W2bT, (int)Kd, (int)Kd);

  auto launch = [&](const GemmP& p, int Z) {
    dim3 g((unsigned)(p.N / BN), (unsigned)(p.M / BM), (unsigned)Z);
    gemm_bf16_wmma<<<g, 256, 0, stream>>>(p);
  };

  // ---- QKV projections (scale folded into Q,K; V stored transposed) ----
  {
    GemmP p{};
    p.A = xb; p.lda = Kd; p.ldb = Kd; p.ldc = NQ;
    p.M = (int)M; p.N = (int)NQ; p.K = (int)Kd; p.Hdiv = 1;
    p.Bt = WqbT; p.Cb = Q;  p.alpha = scale; launch(p, 1);
    p.Bt = WkbT; p.Cb = Kp; p.alpha = scale; launch(p, 1);
    p.Bt = WvbT; p.Cb = Vt; p.alpha = 1.f; p.ctrans = 1; p.ldc = M;
    launch(p, 1);
  }
  // ---- scores: S[b,h] = Qh @ Kh^T (Kp already [keys][d]; causal skip) ----
  {
    GemmP p{};
    p.A = Q;   p.lda = NQ; p.sA1 = T * NQ; p.sA2 = Kd;
    p.Bt = Kp; p.ldb = NQ; p.sB1 = T * NQ; p.sB2 = Kd;
    p.Cf = S;  p.ldc = T;  p.sC1 = Hh * T * T; p.sC2 = T * T;
    p.M = (int)T; p.N = (int)T; p.K = (int)Kd; p.Hdiv = (int)Hh;
    p.alpha = 1.f; p.causal = 1;
    launch(p, (int)(Bb * Hh));
  }
  // ---- softmax with causal mask -> P (bf16) ----
  softmax_causal<<<(unsigned)(Bb * Hh * T), 256, 0, stream>>>(S, P, (int)T);
  // ---- PV: AO[b, :, h*K:(h+1)*K] = P[b,h] @ Vh  (B^T = Vt) ----
  {
    GemmP p{};
    p.A = P;   p.lda = T; p.sA1 = Hh * T * T; p.sA2 = T * T;
    p.Bt = Vt; p.ldb = M; p.sB1 = T;          p.sB2 = Kd * M;
    p.Cb = AO; p.ldc = NQ; p.sC1 = T * NQ;    p.sC2 = Kd;
    p.M = (int)T; p.N = (int)Kd; p.K = (int)T; p.Hdiv = (int)Hh;
    p.alpha = 1.f;
    launch(p, (int)(Bb * Hh));
  }
  // ---- unify heads: xres = x + AO @ Wu + bu  (also bf16 copy for FFN) ----
  {
    GemmP p{};
    p.A = AO; p.lda = NQ; p.Bt = WubT; p.ldb = NQ;
    p.Cf = xres; p.Cb = xrb; p.ldc = Kd;
    p.bias = bu; p.R = x; p.ldr = Kd;
    p.M = (int)M; p.N = (int)Kd; p.K = (int)NQ; p.Hdiv = 1; p.alpha = 1.f;
    launch(p, 1);
  }
  // ---- FFN layer 1: h1 = relu(xres @ W1 + b1) ----
  {
    GemmP p{};
    p.A = xrb; p.lda = Kd; p.Bt = W1bT; p.ldb = Kd;
    p.Cb = h1; p.ldc = Kd; p.bias = b1; p.relu = 1;
    p.M = (int)M; p.N = (int)Kd; p.K = (int)Kd; p.Hdiv = 1; p.alpha = 1.f;
    launch(p, 1);
  }
  // ---- FFN layer 2 + residual: out = xres + h1 @ W2 + b2 ----
  {
    GemmP p{};
    p.A = h1; p.lda = Kd; p.Bt = W2bT; p.ldb = Kd;
    p.Cf = (float*)d_out; p.ldc = Kd; p.bias = b2; p.R = xres; p.ldr = Kd;
    p.M = (int)M; p.N = (int)Kd; p.K = (int)Kd; p.Hdiv = 1; p.alpha = 1.f;
    launch(p, 1);
  }
}
